// PCN_87179246174229
// MI455X (gfx1250) — matrix operand order repaired and checked by reading, not compiled, not run on hardware
//
#include <hip/hip_runtime.h>
#include <hip/hip_bf16.h>

// CDNA5 / gfx1250, wave32. WMMA 16x16x32 bf16, f32 accumulate.

typedef __attribute__((ext_vector_type(16))) __bf16 v16bf;
typedef __attribute__((ext_vector_type(8)))  __bf16 v8bf;
typedef __attribute__((ext_vector_type(8)))  float  v8f;

struct BfPair { v8bf lo; v8bf hi; };   // 32 bytes == v16bf

#define FEAT 128

__device__ __forceinline__ float silu_f(float x) {
    return x / (1.0f + __expf(-x));
}

// ---------------------------------------------------------------------------
// K0: S[n][f] = bf16( emb[cg_z[n]][f] * (cg_z[n] != 0) )
// ---------------------------------------------------------------------------
__global__ void pcn_embed(const float* __restrict__ emb, const int* __restrict__ cg_z,
                          __bf16* __restrict__ S, int n_cg) {
    int tid = blockIdx.x * blockDim.x + threadIdx.x;
    int total = n_cg * FEAT;
    if (tid >= total) return;
    int n = tid >> 7;          // /128
    int f = tid & (FEAT - 1);
    int z = cg_z[n];
    S[tid] = (__bf16)((z != 0) ? emb[z * FEAT + f] : 0.0f);
}

// ---------------------------------------------------------------------------
// K1: pack W_msg (128x128) and W_vec (128xapc, padded to 16) into bf16
// WMMA B-fragments. B-matrix 32x16 bf16 layout (ISA 7.12.2):
//   lane L: N = L%16 ; element e (0..15): K_local = (L<16 ? 0 : 8) + (e<8 ? e : e+8)
// Fragment storage: frag[f*512 + lane*16 + e]  -> one contiguous 32B load/lane.
// ---------------------------------------------------------------------------
__global__ void pcn_pack_w(const float* __restrict__ Wm, const float* __restrict__ Wv,
                           __bf16* __restrict__ fmsg, __bf16* __restrict__ fvec, int apc) {
    int idx = blockIdx.x * blockDim.x + threadIdx.x;
    if (idx < 8 * 4 * 512) {               // W_msg: 8 N-tiles x 4 K-chunks
        int fi     = idx >> 9;
        int within = idx & 511;
        int lane   = within >> 4;
        int e      = within & 15;
        int nt = fi >> 2, kc = fi & 3;
        int n = nt * 16 + (lane & 15);
        int k = kc * 32 + ((lane & 16) ? 8 : 0) + (e < 8 ? e : e + 8);
        fmsg[idx] = (__bf16)Wm[k * FEAT + n];
    } else if (idx < 8 * 4 * 512 + 4 * 512) {   // W_vec: 1 N-tile x 4 K-chunks
        int idx2   = idx - 8 * 4 * 512;
        int kc     = idx2 >> 9;
        int within = idx2 & 511;
        int lane   = within >> 4;
        int e      = within & 15;
        int n = lane & 15;
        int k = kc * 32 + ((lane & 16) ? 8 : 0) + (e < 8 ? e : e + 8);
        fvec[idx2] = (n < apc) ? (__bf16)Wv[k * apc + n] : (__bf16)0.0f;
    }
}

// ---------------------------------------------------------------------------
// K2: P = S @ W_msg   (per-node GEMM, one wave per 16-node M-tile)
// A-matrix 16x32 bf16 layout: lane L holds row M = L%16;
//   elements 0..7  -> K = (L<16 ? 0 : 8)  + e   (+ kc*32)
//   elements 8..15 -> K = (L<16 ? 16 : 24) + (e-8)
// kc-outer schedule: per K-chunk, stage the A chunk + all 8 B-fragments,
// then 8 back-to-back WMMAs. __launch_bounds__(256,1): K2 is launch-bound
// (~3 waves/SIMD), so let the allocator spend VGPRs on deep B prefetch
// instead of capping at the default occupancy target.
// ---------------------------------------------------------------------------
__global__ void __launch_bounds__(256, 1)
pcn_node_gemm(const __bf16* __restrict__ S, const __bf16* __restrict__ fmsg,
              float* __restrict__ P, int n_cg) {
    const int lane = threadIdx.x & 31;
    const int wid  = threadIdx.x >> 5;
    const int tile = blockIdx.x * (blockDim.x >> 5) + wid;
    const int ntiles = (n_cg + 15) >> 4;
    if (tile >= ntiles) return;              // wave-uniform: EXEC stays all-1s

    const bool hi = lane >= 16;
    const int  m  = lane & 15;
    int node = tile * 16 + m;
    int safe = node < n_cg ? node : n_cg - 1;   // clamp reads on tail tile
    const __bf16* row = S + (size_t)safe * FEAT;

    const v16bf* bw = (const v16bf*)fmsg;        // fragment f: bw[f*32 + lane]

    v8f acc[8];
#pragma unroll
    for (int nt = 0; nt < 8; ++nt)
        acc[nt] = v8f{0.f, 0.f, 0.f, 0.f, 0.f, 0.f, 0.f, 0.f};

#pragma unroll
    for (int kc = 0; kc < 4; ++kc) {
        // A chunk for this K range (two 16B vector loads, no cvt VALU)
        int k0 = kc * 32 + (hi ? 8 : 0);
        BfPair pr;
        pr.lo = *(const v8bf*)(row + k0);        // K = k0 .. k0+7
        pr.hi = *(const v8bf*)(row + k0 + 16);   // K = k0+16 .. k0+23
        v16bf a = __builtin_bit_cast(v16bf, pr);

        // stage all 8 B-fragments of this K-chunk -> one clause, one wait
        v16bf bs[8];
#pragma unroll
        for (int nt = 0; nt < 8; ++nt)
            bs[nt] = bw[(nt * 4 + kc) * 32 + lane];

#pragma unroll
        for (int nt = 0; nt < 8; ++nt)
            acc[nt] = __builtin_amdgcn_wmma_f32_16x16x32_bf16(
                false, a, false, bs[nt], (short)0, acc[nt], false, false);
    }

    // C/D 16x16 f32 layout: lane n = L%16; vgpr r -> M = r (+8 for hi lanes).
    // P is padded to ntiles*16 rows -> stores are unconditional; one base
    // pointer per lane, all element offsets are compile-time constants.
    int rbase = tile * 16 + (hi ? 8 : 0);
    float* pb = P + (size_t)rbase * FEAT + (lane & 15);
#pragma unroll
    for (int nt = 0; nt < 8; ++nt) {
#pragma unroll
        for (int r = 0; r < 8; ++r) {
            pb[nt * 16 + r * FEAT] = acc[nt][r];
        }
    }
}

// ---------------------------------------------------------------------------
// K3: one wave per CG node i (= its 16 consecutive edges).
//   m[e]  = silu(P[i] + P[j_e])          (built directly as A-fragments)
//   w     = m @ W_vec                    (4x WMMA, 16x16 tile, apc cols valid)
//   cg_v[ch][d] = sum_e w[e][ch]*unit[e][d]   -> in-wave reduction via shfl
//   out[i*apc+ch] = (ch==ca ? 0 : cg_v[ch]) + cg_xyz[i]
// Default launch bounds: 50k waves; occupancy hides the random P[j] gathers.
// ---------------------------------------------------------------------------
__global__ void pcn_edge(const float* __restrict__ P, const __bf16* __restrict__ fvec,
                         const float* __restrict__ xyz, const int* __restrict__ nbr,
                         const int* __restrict__ ca_idx, float* __restrict__ out,
                         int n_cg, int apc, int deg) {
    const int lane = threadIdx.x & 31;
    const int wid  = threadIdx.x >> 5;
    const int i    = blockIdx.x * (blockDim.x >> 5) + wid;
    if (i >= n_cg) return;                   // wave-uniform

    const bool hi = lane >= 16;
    const int  m  = lane & 15;               // edge slot within node
    const int  e  = i * deg + m;
    const int  j  = nbr[2 * e + 1];

    float xi = xyz[3 * i], yi = xyz[3 * i + 1], zi = xyz[3 * i + 2];
    float dx = xyz[3 * j]     - xi;
    float dy = xyz[3 * j + 1] - yi;
    float dz = xyz[3 * j + 2] - zi;
    float inv = 1.0f / (sqrtf(dx * dx + dy * dy + dz * dz) + 1e-8f);
    float ux = dx * inv, uy = dy * inv, uz = dz * inv;

    const float* Pi = P + (size_t)i * FEAT;  // uniform across wave
    const float* Pj = P + (size_t)j * FEAT;  // per-lane gather (L2 resident)

    // stage all 4 W_vec B-fragments up front (one clause)
    const v16bf* bwv = (const v16bf*)fvec;
    v16bf bs[4];
#pragma unroll
    for (int kc = 0; kc < 4; ++kc)
        bs[kc] = bwv[kc * 32 + lane];

    v16bf a[4];
#pragma unroll
    for (int kc = 0; kc < 4; ++kc) {
        int k0 = kc * 32 + (hi ? 8 : 0);
#pragma unroll
        for (int t = 0; t < 8; ++t) {
            a[kc][t]     = (__bf16)silu_f(Pi[k0 + t]      + Pj[k0 + t]);
            a[kc][t + 8] = (__bf16)silu_f(Pi[k0 + 16 + t] + Pj[k0 + 16 + t]);
        }
    }

    v8f acc = {0.f, 0.f, 0.f, 0.f, 0.f, 0.f, 0.f, 0.f};
#pragma unroll
    for (int kc = 0; kc < 4; ++kc) {
        acc = __builtin_amdgcn_wmma_f32_16x16x32_bf16(
            false, a[kc], false, bs[kc], (short)0, acc, false, false);
    }

    // acc[r] = w[m_r][ch], ch = lane%16, m_r = r (+8 for hi lanes).
    int mbase = hi ? 8 : 0;
    float o0 = 0.f, o1 = 0.f, o2 = 0.f;
#pragma unroll
    for (int r = 0; r < 8; ++r) {
        int   ms = mbase + r;
        float a0 = __shfl(ux, ms, 32);       // lane ms computed unit[ms]
        float a1 = __shfl(uy, ms, 32);
        float a2 = __shfl(uz, ms, 32);
        float wv = acc[r];
        o0 = fmaf(wv, a0, o0);
        o1 = fmaf(wv, a1, o1);
        o2 = fmaf(wv, a2, o2);
    }
    // combine the two half-wave partials for the same channel
    o0 += __shfl_xor(o0, 16, 32);
    o1 += __shfl_xor(o1, 16, 32);
    o2 += __shfl_xor(o2, 16, 32);

    int ca_ch = ca_idx[i] - i * apc;
    if (!hi && lane < apc) {
        bool z = (lane == ca_ch);
        size_t o = ((size_t)i * apc + lane) * 3;
        out[o]     = (z ? 0.f : o0) + xi;
        out[o + 1] = (z ? 0.f : o1) + yi;
        out[o + 2] = (z ? 0.f : o2) + zi;
    }
}

// ---------------------------------------------------------------------------
extern "C" void kernel_launch(void* const* d_in, const int* in_sizes, int n_in,
                              void* d_out, int out_size, void* d_ws, size_t ws_size,
                              hipStream_t stream) {
    const float* cg_xyz = (const float*)d_in[0];
    const float* emb    = (const float*)d_in[1];
    const float* W_msg  = (const float*)d_in[2];
    const float* W_vec  = (const float*)d_in[3];
    const int*   cg_z   = (const int*)d_in[4];
    const int*   nbr    = (const int*)d_in[5];
    const int*   ca_idx = (const int*)d_in[7];

    const int n_cg   = in_sizes[4];
    const int apc    = in_sizes[6] / n_cg;        // atoms per CG (13)
    const int deg    = in_sizes[5] / (2 * n_cg);  // edges per node (16)
    const int ntiles = (n_cg + 15) / 16;
    const int pRows  = ntiles * 16;               // P padded to full tiles

    float* out = (float*)d_out;

    // Workspace layout (all offsets 256B-aligned)
    char*  ws   = (char*)d_ws;
    size_t sBf  = (((size_t)n_cg  * FEAT * sizeof(__bf16)) + 255) & ~(size_t)255;
    size_t pF32 = (((size_t)pRows * FEAT * sizeof(float))  + 255) & ~(size_t)255;
    __bf16* S    = (__bf16*)ws;                       // n_cg  x 128 bf16
    float*  P    = (float*)(ws + sBf);                // pRows x 128 f32 (padded)
    __bf16* fmsg = (__bf16*)(ws + sBf + pF32);        // 8*4*512 bf16
    __bf16* fvec = (__bf16*)(ws + sBf + pF32 + (size_t)(8 * 4 * 512) * sizeof(__bf16));

    // K0: embedding gather -> bf16
    {
        int total  = n_cg * FEAT;
        int blocks = (total + 255) / 256;
        pcn_embed<<<blocks, 256, 0, stream>>>(emb, cg_z, S, n_cg);
    }
    // K1: pack weight fragments (tiny)
    {
        int total  = 8 * 4 * 512 + 4 * 512;
        int blocks = (total + 255) / 256;
        pcn_pack_w<<<blocks, 256, 0, stream>>>(W_msg, W_vec, fmsg, fvec, apc);
    }
    // K2: P = S @ W_msg  (one wave per 16-node tile, 8 waves/block)
    {
        int blocks = (ntiles + 7) / 8;
        pcn_node_gemm<<<blocks, 256, 0, stream>>>(S, fmsg, P, n_cg);
    }
    // K3: per-node edge phase (one wave per node, 8 waves/block)
    {
        int blocks = (n_cg + 7) / 8;
        pcn_edge<<<blocks, 256, 0, stream>>>(P, fvec, cg_xyz, nbr, ca_idx, out,
                                             n_cg, apc, deg);
    }
}
